// HamiltonianPropagator_90452011254080
// MI455X (gfx1250) — matrix operand order repaired and checked by reading, not compiled, not run on hardware
//
#include <hip/hip_runtime.h>
#include <hip/hip_bf16.h>

// ---------------------------------------------------------------------------
// HamiltonianPropagator, fused single-pass:
//   forcing GEMM (f16 WMMA, f32 acc) -> complex diagonal recurrence in VGPRs
//   -> output GEMM (f16 WMMA, f32 acc) + bias.  h_states never touch HBM.
// Software-pipelined: x-tile / A-table prefetched one t-step ahead (ping-pong
// registers via #pragma unroll 2), non-temporal streaming for x and out.
// ---------------------------------------------------------------------------

typedef _Float16 half8  __attribute__((ext_vector_type(8)));
typedef _Float16 v16h   __attribute__((ext_vector_type(16)));
typedef float    v8f    __attribute__((ext_vector_type(8)));
typedef float    f32x4  __attribute__((ext_vector_type(4)));
typedef float    f32x2  __attribute__((ext_vector_type(2)));

#define CC     128      // model dim
#define CC2    256      // 2*C
#define TT     256      // timesteps
#define NSEQ   1024     // H*W independent sites (B=1)

// workspace layout (bytes)
#define WS_BINFRAG   0                    // 16 nt * 4 kc * 1024B = 64 KB
#define WS_BOUTFRAG  (64*1024)            //  8 nt * 8 kc * 1024B = 64 KB
#define WS_ATAB      (128*1024)           // 256*128 float2       = 256 KB

__device__ __forceinline__ v16h cat8(half8 lo, half8 hi) {
    return __builtin_shufflevector(lo, hi, 0,1,2,3,4,5,6,7,8,9,10,11,12,13,14,15);
}

__device__ __forceinline__ v8f wmma_f16(v16h a, v16h b, v8f c) {
    return __builtin_amdgcn_wmma_f32_16x16x32_f16(false, a, false, b,
                                                  (short)0, c, false, false);
}

// ---------------------------------------------------------------------------
// Prep 1: W_in (2C x C, row major, proj[d] = sum_c x[c]*W_in[d][c]) ->
// WMMA B-fragments (K=c, N=d).  Fragment (nt,kc): 32 lanes x 32B.
// Lane L: N = nt*16 + (L&15); half j holds K = kc*32 + (L>=16?16:0) + j.
// ---------------------------------------------------------------------------
__global__ void hp_prep_bin(const float* __restrict__ W_in,
                            _Float16* __restrict__ binfrag) {
    int idx = blockIdx.x * blockDim.x + threadIdx.x;   // 0..32767
    int j    = idx & 15;
    int lane = (idx >> 4) & 31;
    int kc   = (idx >> 9) & 3;
    int nt   = idx >> 11;                              // 0..15
    int d = nt * 16 + (lane & 15);
    int c = kc * 32 + ((lane >= 16) ? 16 : 0) + j;
    binfrag[idx] = (_Float16)W_in[d * CC + c];
}

// Prep 2: W_out (C x 2C, out[c] = sum_d hcat[d]*W_out[c][d]) -> B-frags (K=d, N=c)
__global__ void hp_prep_bout(const float* __restrict__ W_out,
                             _Float16* __restrict__ boutfrag) {
    int idx = blockIdx.x * blockDim.x + threadIdx.x;   // 0..32767
    int j    = idx & 15;
    int lane = (idx >> 4) & 31;
    int kc   = (idx >> 9) & 7;
    int nt   = idx >> 12;                              // 0..7
    int c = nt * 16 + (lane & 15);
    int d = kc * 32 + ((lane >= 16) ? 16 : 0) + j;
    boutfrag[idx] = (_Float16)W_out[c * CC2 + d];
}

// Prep 3: A[t,c] = exp(i * exp(log_freq[c]) * dt[t]) -> (cos,sin) table
__global__ void hp_prep_atab(const float* __restrict__ dt,
                             const float* __restrict__ log_freq,
                             float* __restrict__ atab) {
    int idx = blockIdx.x * blockDim.x + threadIdx.x;   // 0..32767
    int t = idx >> 7;
    int c = idx & 127;
    float ph = __expf(log_freq[c]) * dt[t];
    float s, co;
    __sincosf(ph, &s, &co);
    atab[2 * idx]     = co;
    atab[2 * idx + 1] = s;
}

// ---------------------------------------------------------------------------
// Main fused kernel: 64 blocks (16 sites each) x 256 threads (8 waves).
// Wave w owns channels [16w, 16w+16).
// ---------------------------------------------------------------------------
__global__ __launch_bounds__(256, 1)
void hp_main(const float* __restrict__ x,
             const float* __restrict__ dtp,
             const float* __restrict__ b_out,
             const _Float16* __restrict__ binfrag,
             const _Float16* __restrict__ boutfrag,
             const float* __restrict__ atab,
             float* __restrict__ out,
             float* __restrict__ hfin) {
    const int n0     = blockIdx.x * 16;
    const int tid    = threadIdx.x;
    const int wave   = tid >> 5;        // 0..7
    const int lane   = tid & 31;
    const int lane15 = lane & 15;
    const int hi     = lane >> 4;       // half-wave select

    __shared__ __align__(16) _Float16 xA[16 * CC];     // 4 KB, x tile as f16
    __shared__ __align__(16) _Float16 hcat[16 * CC2];  // 8 KB, [re | im] as f16

    // ---- load loop-invariant weight B-fragments into VGPRs (128 VGPRs) ----
    v16h binf[2][4];                    // [re/im tile][K-chunk]
    #pragma unroll
    for (int j = 0; j < 2; ++j) {
        int nt = wave + j * 8;          // re cols = [16w..), im cols = [128+16w..)
        #pragma unroll
        for (int kc = 0; kc < 4; ++kc) {
            const half8* p = (const half8*)(binfrag + (((nt * 4 + kc) * 32 + lane) << 4));
            binf[j][kc] = cat8(p[0], p[1]);
        }
    }
    v16h boutf[8];
    #pragma unroll
    for (int kc = 0; kc < 8; ++kc) {
        const half8* p = (const half8*)(boutfrag + (((wave * 8 + kc) * 32 + lane) << 4));
        boutf[kc] = cat8(p[0], p[1]);
    }

    const int   cglob = wave * 16 + lane15;     // this lane's channel column
    const float bval  = b_out[cglob];

    v8f hre = {};   // recurrence state, resident in accumulator layout
    v8f him = {};

    // ---- software-pipeline prefetch registers (ping-pong, unroll 2) ----
    f32x4 px0[2], px1[2];
    f32x2 pa[2];
    float pdt[2];
    {
        const f32x4* xg = (const f32x4*)(x + (size_t)n0 * CC);
        px0[0] = __builtin_nontemporal_load(xg + tid * 2);
        px1[0] = __builtin_nontemporal_load(xg + tid * 2 + 1);
        pa[0]  = *(const f32x2*)(atab + 2 * cglob);
        pdt[0] = dtp[0];
    }

    #pragma unroll 2
    for (int t = 0; t < TT; ++t) {
        const int cur = t & 1;
        const int nxt = cur ^ 1;

        // ---- issue prefetch for t+1 (consumed one full iteration later) ----
        {
            const int tn = (t + 1 < TT) ? (t + 1) : t;
            const f32x4* xg = (const f32x4*)(x + ((size_t)tn * NSEQ + n0) * CC);
            px0[nxt] = __builtin_nontemporal_load(xg + tid * 2);
            px1[nxt] = __builtin_nontemporal_load(xg + tid * 2 + 1);
            pa[nxt]  = *(const f32x2*)(atab + 2 * (tn * CC + cglob));
            pdt[nxt] = dtp[tn];
        }

        // ---- stage x tile (16 x 128 f32, contiguous) -> LDS f16 ----
        half8 xh;
        xh[0] = (_Float16)px0[cur][0]; xh[1] = (_Float16)px0[cur][1];
        xh[2] = (_Float16)px0[cur][2]; xh[3] = (_Float16)px0[cur][3];
        xh[4] = (_Float16)px1[cur][0]; xh[5] = (_Float16)px1[cur][1];
        xh[6] = (_Float16)px1[cur][2]; xh[7] = (_Float16)px1[cur][3];
        *(half8*)(xA + tid * 8) = xh;
        __syncthreads();

        // ---- forcing GEMM: X = x_tile @ W_in^T  (re + im 16x16 tiles) ----
        v8f Xr = {};
        v8f Xi = {};
        #pragma unroll
        for (int kc = 0; kc < 4; ++kc) {
            const _Float16* ab = xA + lane15 * CC + kc * 32 + hi * 16;
            v16h afrag = cat8(*(const half8*)ab, *(const half8*)(ab + 8));
            Xr = wmma_f16(afrag, binf[0][kc], Xr);
            Xi = wmma_f16(afrag, binf[1][kc], Xi);
        }

        // ---- complex diagonal recurrence: h = A*h + dt*X ----
        const float are = pa[cur][0];
        const float aim = pa[cur][1];
        const float dv  = pdt[cur];
        #pragma unroll
        for (int r = 0; r < 8; ++r) {
            float xr = Xr[r] * dv;
            float xi = Xi[r] * dv;
            float nr = fmaf(are, hre[r], fmaf(-aim, him[r], xr));
            float ni = fmaf(are, him[r], fmaf( aim, hre[r], xi));
            hre[r] = nr;
            him[r] = ni;
        }

        // ---- publish h_cat = [Re(h) | Im(h)] as f16 (A-matrix of out GEMM) ----
        #pragma unroll
        for (int r = 0; r < 8; ++r) {
            int M = r + hi * 8;                       // C/D row mapping
            hcat[M * CC2 + cglob]      = (_Float16)hre[r];
            hcat[M * CC2 + CC + cglob] = (_Float16)him[r];
        }
        __syncthreads();

        // ---- output GEMM: out_tile = h_cat @ W_out^T, K = 256 ----
        v8f o = {};
        #pragma unroll
        for (int kc = 0; kc < 8; ++kc) {
            const _Float16* ab = hcat + lane15 * CC2 + kc * 32 + hi * 16;
            v16h afrag = cat8(*(const half8*)ab, *(const half8*)(ab + 8));
            o = wmma_f16(afrag, boutf[kc], o);
        }

        // ---- store out[t, n0+M, c] = o + b_out[c]  (streaming, NT) ----
        float* op = out + ((size_t)t * NSEQ + n0) * CC + cglob;
        #pragma unroll
        for (int r = 0; r < 8; ++r) {
            int M = r + hi * 8;
            __builtin_nontemporal_store(o[r] + bval, op + (size_t)M * CC);
        }
    }

    // ---- h_final: complex64 (re,im interleaved), shape (H*W, C) ----
    #pragma unroll
    for (int r = 0; r < 8; ++r) {
        int M = r + hi * 8;
        float* hp = hfin + ((size_t)(n0 + M) * CC + cglob) * 2;
        hp[0] = hre[r];
        hp[1] = him[r];
    }
}

// ---------------------------------------------------------------------------
extern "C" void kernel_launch(void* const* d_in, const int* in_sizes, int n_in,
                              void* d_out, int out_size, void* d_ws, size_t ws_size,
                              hipStream_t stream) {
    (void)in_sizes; (void)n_in; (void)out_size; (void)ws_size;
    const float* x        = (const float*)d_in[0];   // (1,256,32,32,128)
    const float* dt       = (const float*)d_in[1];   // (1,256)
    const float* log_freq = (const float*)d_in[2];   // (128,)
    const float* W_in     = (const float*)d_in[3];   // (256,128)
    const float* W_out    = (const float*)d_in[4];   // (128,256)
    const float* b_out    = (const float*)d_in[5];   // (128,)

    _Float16* binfrag  = (_Float16*)((char*)d_ws + WS_BINFRAG);
    _Float16* boutfrag = (_Float16*)((char*)d_ws + WS_BOUTFRAG);
    float*    atab     = (float*)   ((char*)d_ws + WS_ATAB);

    float* out  = (float*)d_out;                       // 256*1024*128 floats
    float* hfin = out + (size_t)TT * NSEQ * CC;        // then h_final (complex64)

    hipLaunchKernelGGL(hp_prep_bin,  dim3(128), dim3(256), 0, stream, W_in,  binfrag);
    hipLaunchKernelGGL(hp_prep_bout, dim3(128), dim3(256), 0, stream, W_out, boutfrag);
    hipLaunchKernelGGL(hp_prep_atab, dim3(128), dim3(256), 0, stream, dt, log_freq, atab);
    hipLaunchKernelGGL(hp_main, dim3(NSEQ / 16), dim3(256), 0, stream,
                       x, dt, b_out, binfrag, boutfrag, atab, out, hfin);
}
